// MagGatedAttention_6708738916554
// MI455X (gfx1250) — matrix operand order repaired and checked by reading, not compiled, not run on hardware
//
#include <hip/hip_runtime.h>

typedef _Float16 f16;
typedef __attribute__((ext_vector_type(16))) _Float16 v16h;
typedef __attribute__((ext_vector_type(8)))  _Float16 v8h;
typedef __attribute__((ext_vector_type(4)))  _Float16 v4h;
typedef __attribute__((ext_vector_type(8)))  float    v8f;
typedef __attribute__((ext_vector_type(4)))  float    v4f;
typedef __attribute__((ext_vector_type(4)))  unsigned int v4u;
typedef __attribute__((ext_vector_type(4)))  int      v4i;
typedef __attribute__((ext_vector_type(8)))  int      v8i;

#if defined(__has_builtin)
#if __has_builtin(__builtin_amdgcn_tensor_load_to_lds)
#define HAS_TDM 1
#endif
#endif

namespace {

constexpr int Bb = 2, Ss = 1024, Hh = 4096, NH = 32, NKV = 8, Dd = 128;
constexpr int NREP = NH / NKV;
constexpr float SCALE  = 0.08838834764831845f;   // 1/sqrt(128)
constexpr float NEGBIG = -1e9f;

// A-fragment (16x32, M x K): lane holds row (lane&15); K groups
// {0..7, 16..23} for lanes 0-15, {8..15, 24..31} for lanes 16-31.
__device__ __forceinline__ v16h load_fragA(const f16* base, int ldk, int lane) {
    const int row = lane & 15;
    const int lo  = (lane & 16) ? 8 : 0;
    const f16* p = base + row * ldk + lo;
    union { v16h v; v8h h[2]; } u;
    u.h[0] = *(const v8h*)(p);
    u.h[1] = *(const v8h*)(p + 16);
    return u.v;
}

// B-fragment (32x16, K x N) from LDS stored as [n][k] (k contiguous):
// lane holds column (lane&15); K groups {0..7, 16..23} (lanes 0-15) / {8..15, 24..31} (lanes 16-31).
__device__ __forceinline__ v16h load_fragB(const f16* base, int ldk, int lane) {
    const int col = lane & 15;
    const int off = (lane & 16) ? 8 : 0;
    const f16* p = base + col * ldk + off;
    union { v16h v; v8h h[2]; } u;
    u.h[0] = *(const v8h*)(p);
    u.h[1] = *(const v8h*)(p + 16);
    return u.v;
}

__device__ __forceinline__ v8f wmma16(v16h a, v16h b, v8f c) {
    return __builtin_amdgcn_wmma_f32_16x16x32_f16(false, a, false, b, (short)0, c,
                                                  false, false);
}

#if defined(HAS_TDM)
// TDM: async-load a 64-row x 256-byte row-major tile (row stride 256B in
// global) into LDS with a 272B padded row stride (pad 4 DWORDs after every
// 64 DWORDs). data_size = 4B, tile = 64x64 DWORDs.
__device__ __forceinline__ void tdm_load_tile_64x256(unsigned lds_off, const void* g) {
    const unsigned long long ga = (unsigned long long)(size_t)g;
    v4u g0;
    g0[0] = 1u;                                    // count=1 (valid), load, user
    g0[1] = lds_off;                               // LDS destination byte addr
    g0[2] = (unsigned)ga;                          // global_addr[31:0]
    g0[3] = (unsigned)((ga >> 32) & 0x01FFFFFFu)   // global_addr[56:32]
          | (2u << 30);                            // type = 2 (image)
    v8i g1;
    g1[0] = (int)((2u << 16)      // data_size = 4B
                | (1u << 20)      // pad_enable
                | (5u << 22)      // pad_interval: 64 DWORDs (256B)
                | (3u << 25));    // pad_amount:   4 DWORDs (16B)
    g1[1] = (int)(64u << 16);     // tensor_dim0 = 64 DW  (bits[63:48])
    g1[2] = (int)(64u << 16);     // tensor_dim1 = 64 rows (bits[95:80])
    g1[3] = (int)(64u << 16);     // tile_dim0   = 64 DW  (bits[127:112])
    g1[4] = 64;                   // tile_dim1   = 64 rows (bits[143:128])
    g1[5] = 64;                   // tensor_dim0_stride = 64 DW (bits[191:160])
    g1[6] = 0;
    g1[7] = 0;
    const v4i z4 = {0, 0, 0, 0};
#if defined(__clang_major__) && __clang_major__ >= 23
    const v8i z8 = {0, 0, 0, 0, 0, 0, 0, 0};
    __builtin_amdgcn_tensor_load_to_lds(g0, g1, z4, z4, z8, 0);
#else
    __builtin_amdgcn_tensor_load_to_lds(g0, g1, z4, z4, 0);
#endif
}
#endif

} // namespace

// ---------------------------------------------------------------------------
// C[M,N] = A[M,K] @ W[K,N], fp32 in/out, f16 WMMA core.
// Block = 128 threads (4 waves), 64x64 tile, 64-deep K steps, double-buffered
// LDS: global loads for step i+1 issue before the WMMAs of step i; one
// barrier per k-step.
// ---------------------------------------------------------------------------
__global__ __launch_bounds__(128)
void gemm_xw_wmma(const float* __restrict__ A, const float* __restrict__ W,
                  float* __restrict__ C, int M, int N, int K) {
    __shared__ __align__(16) f16 As[2][64][72];   // [m][k], stride 144B
    __shared__ __align__(16) f16 Bs[2][64][72];   // [n][k], stride 144B

    const int tid  = threadIdx.x;
    const int lane = tid & 31;
    const int wave = tid >> 5;
    const int wm = (wave >> 1) * 32;
    const int wn = (wave & 1) * 32;
    const int m0 = blockIdx.y * 64;
    const int n0 = blockIdx.x * 64;

    // per-thread staging coordinates
    const int ar = tid >> 4, ac4 = (tid & 15) * 4;   // A: 8 rows apart per it
    const int bk = tid >> 4, bn4 = (tid & 15) * 4;   // B: 8 k-rows apart per it

    const v8f vzero = {};
    v8f acc[2][2] = {{vzero, vzero}, {vzero, vzero}};

    // ---- prologue: stage k0 = 0 into buffer 0
    #pragma unroll
    for (int it = 0; it < 8; ++it) {
        const int r = ar + it * 8;
        const v4f f = *(const v4f*)&A[(size_t)(m0 + r) * K + ac4];
        const v4h h = {(f16)f[0], (f16)f[1], (f16)f[2], (f16)f[3]};
        *(v4h*)&As[0][r][ac4] = h;
    }
    #pragma unroll
    for (int it = 0; it < 8; ++it) {
        const int kk = bk + it * 8;
        const v4f f = *(const v4f*)&W[(size_t)kk * N + (n0 + bn4)];
        Bs[0][bn4 + 0][kk] = (f16)f[0];
        Bs[0][bn4 + 1][kk] = (f16)f[1];
        Bs[0][bn4 + 2][kk] = (f16)f[2];
        Bs[0][bn4 + 3][kk] = (f16)f[3];
    }
    __syncthreads();

    int cur = 0;
    for (int k0 = 0; k0 < K; k0 += 64) {
        const bool has_next = (k0 + 64 < K);
        v4f fa[8], fb[8];
        if (has_next) {
            // issue next tile's global loads before doing any math
            #pragma unroll
            for (int it = 0; it < 8; ++it)
                fa[it] = *(const v4f*)&A[(size_t)(m0 + ar + it * 8) * K + (k0 + 64 + ac4)];
            #pragma unroll
            for (int it = 0; it < 8; ++it)
                fb[it] = *(const v4f*)&W[(size_t)(k0 + 64 + bk + it * 8) * N + (n0 + bn4)];
            if (k0 + 128 < K) {   // global_prefetch_b8 two steps ahead
                __builtin_prefetch(&A[(size_t)(m0 + (tid >> 1)) * K + (k0 + 128 + (tid & 1) * 32)], 0, 1);
                __builtin_prefetch(&W[(size_t)(k0 + 128 + (tid & 63)) * N + (n0 + (tid >> 6) * 32)], 0, 1);
            }
        }

        const f16* Ab = &As[cur][0][0];
        const f16* Bc = &Bs[cur][0][0];
        #pragma unroll
        for (int kc = 0; kc < 2; ++kc)
            #pragma unroll
            for (int tm = 0; tm < 2; ++tm) {
                const v16h a = load_fragA(Ab + (wm + tm * 16) * 72 + kc * 32, 72, lane);
                #pragma unroll
                for (int tn = 0; tn < 2; ++tn) {
                    const v16h b = load_fragB(Bc + (wn + tn * 16) * 72 + kc * 32, 72, lane);
                    acc[tm][tn] = wmma16(a, b, acc[tm][tn]);
                }
            }

        if (has_next) {
            const int nxt = cur ^ 1;
            #pragma unroll
            for (int it = 0; it < 8; ++it) {
                const v4h h = {(f16)fa[it][0], (f16)fa[it][1], (f16)fa[it][2], (f16)fa[it][3]};
                *(v4h*)&As[nxt][ar + it * 8][ac4] = h;
            }
            #pragma unroll
            for (int it = 0; it < 8; ++it) {
                const int kk = bk + it * 8;
                Bs[nxt][bn4 + 0][kk] = (f16)fb[it][0];
                Bs[nxt][bn4 + 1][kk] = (f16)fb[it][1];
                Bs[nxt][bn4 + 2][kk] = (f16)fb[it][2];
                Bs[nxt][bn4 + 3][kk] = (f16)fb[it][3];
            }
        }
        __syncthreads();
        cur ^= 1;
    }

    const int ncol = lane & 15;
    const int moff = (lane & 16) ? 8 : 0;
    #pragma unroll
    for (int tm = 0; tm < 2; ++tm)
        #pragma unroll
        for (int tn = 0; tn < 2; ++tn) {
            const int n  = n0 + wn + tn * 16 + ncol;
            const int mb = m0 + wm + tm * 16 + moff;
            #pragma unroll
            for (int r = 0; r < 8; ++r)
                C[(size_t)(mb + r) * N + n] = acc[tm][tn][r];
        }
}

// ---------------------------------------------------------------------------
// RoPE + repack, 4-wide: X[B*S, nh*D] f32 -> Y[(b*nh+h)*S + s][D] f16.
// ---------------------------------------------------------------------------
__global__ __launch_bounds__(256)
void rope_pack4(const float* __restrict__ X, const float* __restrict__ cosT,
                const float* __restrict__ sinT, f16* __restrict__ Y,
                int nheads, int apply_rope, int total4) {
    const int e4 = blockIdx.x * 256 + threadIdx.x;
    if (e4 >= total4) return;
    const int d = (e4 & 31) * 4;          // D/4 = 32 chunks per row
    int t = e4 >> 5;
    const int hh = t % nheads;  t /= nheads;
    const int s  = t % Ss;
    const int b  = t / Ss;
    const size_t inb = (((size_t)b * Ss + s) * nheads + hh) * Dd + d;
    const v4f x = *(const v4f*)&X[inb];
    v4f v = x;
    if (apply_rope) {
        const v4f other = (d < Dd / 2) ? -(*(const v4f*)&X[inb + Dd / 2])
                                       :  (*(const v4f*)&X[inb - Dd / 2]);
        const size_t ci = ((size_t)b * Ss + s) * Dd + d;
        const v4f c  = *(const v4f*)&cosT[ci];
        const v4f sn = *(const v4f*)&sinT[ci];
        v = x * c + other * sn;
    }
    const v4h hv = {(f16)v[0], (f16)v[1], (f16)v[2], (f16)v[3]};
    *(v4h*)&Y[(((size_t)b * nheads + hh) * Ss + s) * Dd + d] = hv;
}

// ---------------------------------------------------------------------------
// Causal flash attention. Block: 128 threads (4 waves), one 64-query tile per
// block; each wave owns 16 query rows. Q/K tiles arrive via the Tensor Data
// Mover (async DMA -> LDS, TENSORcnt), V via a batched transposing copy.
// Writes O as f32 [B*S, NH*D].
// ---------------------------------------------------------------------------
__global__ __launch_bounds__(128)
void flash_attn_wmma(const f16* __restrict__ Qh, const f16* __restrict__ Kh,
                     const f16* __restrict__ Vh, float* __restrict__ O) {
    __shared__ __align__(16) f16 Qs[64][136];   // [q][d]    272B stride
    __shared__ __align__(16) f16 Ks[64][136];   // [key][d]  == B-layout [n][k]
    __shared__ __align__(16) f16 Vt[128][72];   // [d][key]  == B-layout [n][k]
    __shared__ __align__(16) f16 Ps[64][72];    // [q][key]  == A-layout

    const int tid  = threadIdx.x;
    const int lane = tid & 31;
    const int wave = tid >> 5;

    const int qi = blockIdx.x & 15;          // S/64 = 16 query tiles
    const int h  = (blockIdx.x >> 4) & 31;
    const int b  = blockIdx.x >> 9;
    const int hk = h / NREP;

    const f16* Qb = Qh + ((size_t)(b * NH  + h ) * Ss) * Dd;
    const f16* Kb = Kh + ((size_t)(b * NKV + hk) * Ss) * Dd;
    const f16* Vb = Vh + ((size_t)(b * NKV + hk) * Ss) * Dd;

#if defined(HAS_TDM)
    if (wave == 0)
        tdm_load_tile_64x256((unsigned)(size_t)&Qs[0][0], Qb + (size_t)(qi * 64) * Dd);
#else
    #pragma unroll
    for (int it = 0; it < 8; ++it) {
        const int i = tid + it * 128;
        const int r = i >> 4, c8 = (i & 15) * 8;
        *(v8h*)&Qs[r][c8] = *(const v8h*)(Qb + (size_t)(qi * 64 + r) * Dd + c8);
    }
#endif

    const v8f vzero = {};
    v8f o[8];
    #pragma unroll
    for (int dt = 0; dt < 8; ++dt) o[dt] = vzero;
    float mrow[8], lrow[8];
    #pragma unroll
    for (int r = 0; r < 8; ++r) { mrow[r] = -1e30f; lrow[r] = 0.f; }

    const int ncol = lane & 15;
    const int moff = (lane & 16) ? 8 : 0;

    for (int kt = 0; kt <= qi; ++kt) {       // causal: skip tiles above diag
        const int k0 = kt * 64;
#if defined(HAS_TDM)
        if (wave == 0)
            tdm_load_tile_64x256((unsigned)(size_t)&Ks[0][0], Kb + (size_t)k0 * Dd);
#else
        #pragma unroll
        for (int it = 0; it < 8; ++it) {
            const int i = tid + it * 128;
            const int r = i >> 4, c8 = (i & 15) * 8;
            *(v8h*)&Ks[r][c8] = *(const v8h*)(Kb + (size_t)(k0 + r) * Dd + c8);
        }
#endif
        // V tile, transposed into [d][key]: batch loads, then scatter
        {
            const int key = tid >> 4, d8 = (tid & 15) * 8;
            v8h hv[8];
            #pragma unroll
            for (int it = 0; it < 8; ++it)
                hv[it] = *(const v8h*)(Vb + (size_t)(k0 + key + it * 8) * Dd + d8);
            #pragma unroll
            for (int it = 0; it < 8; ++it)
                #pragma unroll
                for (int j = 0; j < 8; ++j)
                    Vt[d8 + j][key + it * 8] = hv[it][j];
        }
#if defined(HAS_TDM)
        if (wave == 0) __builtin_amdgcn_s_wait_tensorcnt(0);
#endif
        __syncthreads();

        // scores: S = Q @ K^T  (this wave: 16 rows x 64 keys)
        v8f sc[4];
        #pragma unroll
        for (int nt = 0; nt < 4; ++nt) sc[nt] = vzero;
        #pragma unroll
        for (int nt = 0; nt < 4; ++nt)
            #pragma unroll
            for (int kc = 0; kc < 4; ++kc) {
                const v16h a  = load_fragA(&Qs[wave * 16][kc * 32], 136, lane);
                const v16h bb = load_fragB(&Ks[nt   * 16][kc * 32], 136, lane);
                sc[nt] = wmma16(a, bb, sc[nt]);
            }

        const bool diag = (kt == qi);
        #pragma unroll
        for (int nt = 0; nt < 4; ++nt)
            #pragma unroll
            for (int r = 0; r < 8; ++r) {
                float s = sc[nt][r] * SCALE;
                if (diag && (k0 + nt * 16 + ncol) > (qi * 64 + wave * 16 + r + moff))
                    s = NEGBIG;
                sc[nt][r] = s;
            }

        // row max: intra-lane over nt, then across the 16-lane half
        float mnew[8];
        #pragma unroll
        for (int r = 0; r < 8; ++r)
            mnew[r] = fmaxf(fmaxf(sc[0][r], sc[1][r]), fmaxf(sc[2][r], sc[3][r]));
        #pragma unroll
        for (int msk = 1; msk < 16; msk <<= 1)
            #pragma unroll
            for (int r = 0; r < 8; ++r)
                mnew[r] = fmaxf(mnew[r], __shfl_xor(mnew[r], msk, 32));

        float alpha[8];
        #pragma unroll
        for (int r = 0; r < 8; ++r) {
            const float mn = fmaxf(mrow[r], mnew[r]);
            alpha[r] = __expf(mrow[r] - mn);
            mrow[r]  = mn;
        }

        float rs[8];
        #pragma unroll
        for (int r = 0; r < 8; ++r) rs[r] = 0.f;
        #pragma unroll
        for (int nt = 0; nt < 4; ++nt)
            #pragma unroll
            for (int r = 0; r < 8; ++r) {
                const float p = __expf(sc[nt][r] - mrow[r]);
                sc[nt][r] = p;
                rs[r] += p;
            }
        #pragma unroll
        for (int msk = 1; msk < 16; msk <<= 1)
            #pragma unroll
            for (int r = 0; r < 8; ++r)
                rs[r] += __shfl_xor(rs[r], msk, 32);
        #pragma unroll
        for (int r = 0; r < 8; ++r) lrow[r] = lrow[r] * alpha[r] + rs[r];
        #pragma unroll
        for (int dt = 0; dt < 8; ++dt)
            #pragma unroll
            for (int r = 0; r < 8; ++r) o[dt][r] *= alpha[r];

        // C-layout -> A-layout via LDS (wave-private rows, no barrier needed)
        #pragma unroll
        for (int nt = 0; nt < 4; ++nt)
            #pragma unroll
            for (int r = 0; r < 8; ++r)
                Ps[wave * 16 + r + moff][nt * 16 + ncol] = (f16)sc[nt][r];

        // O += P @ V  (16x64 @ 64x128)
        #pragma unroll
        for (int dt = 0; dt < 8; ++dt)
            #pragma unroll
            for (int kc = 0; kc < 2; ++kc) {
                const v16h a  = load_fragA(&Ps[wave * 16][kc * 32], 72, lane);
                const v16h bb = load_fragB(&Vt[dt   * 16][kc * 32], 72, lane);
                o[dt] = wmma16(a, bb, o[dt]);
            }
        __syncthreads();
    }

    float inv[8];
    #pragma unroll
    for (int r = 0; r < 8; ++r) inv[r] = 1.0f / lrow[r];
    float* Op = O + ((size_t)(b * Ss + qi * 64)) * (NH * Dd) + h * Dd;
    #pragma unroll
    for (int dt = 0; dt < 8; ++dt)
        #pragma unroll
        for (int r = 0; r < 8; ++r) {
            const int m = wave * 16 + r + moff;
            const int d = dt * 16 + ncol;
            Op[(size_t)m * (NH * Dd) + d] = o[dt][r] * inv[r];
        }
}

// ---------------------------------------------------------------------------
extern "C" void kernel_launch(void* const* d_in, const int* in_sizes, int n_in,
                              void* d_out, int out_size, void* d_ws, size_t ws_size,
                              hipStream_t stream) {
    (void)in_sizes; (void)n_in; (void)out_size; (void)ws_size;
    const float* hs   = (const float*)d_in[0];
    const float* cosT = (const float*)d_in[1];
    const float* sinT = (const float*)d_in[2];
    // d_in[3] = attention_mask: causal mask applied analytically, unused.
    const float* Wq = (const float*)d_in[4];
    const float* Wk = (const float*)d_in[5];
    const float* Wv = (const float*)d_in[6];
    const float* Wo = (const float*)d_in[7];
    float* out = (float*)d_out;

    char* ws = (char*)d_ws;
    // fp32 staging (Q aliased by O after RoPE consumes it)
    float* Qf = (float*)(ws + 0);                    // 32 MB
    float* Kf = (float*)(ws + (size_t)33554432);     //  8 MB
    float* Vf = (float*)(ws + (size_t)41943040);     //  8 MB
    float* Of = Qf;                                  // alias (Qf dead by then)
    // f16 head-major tensors
    f16* Qh = (f16*)(ws + (size_t)50331648);         // 16 MB
    f16* Kh = (f16*)(ws + (size_t)67108864);         //  4 MB
    f16* Vh = (f16*)(ws + (size_t)71303168);         //  4 MB

    const int M = Bb * Ss;                           // 2048
    dim3 blk(128);

    gemm_xw_wmma<<<dim3(Hh / 64, M / 64), blk, 0, stream>>>(hs, Wq, Qf, M, NH  * Dd, Hh);
    gemm_xw_wmma<<<dim3((NKV * Dd) / 64, M / 64), blk, 0, stream>>>(hs, Wk, Kf, M, NKV * Dd, Hh);
    gemm_xw_wmma<<<dim3((NKV * Dd) / 64, M / 64), blk, 0, stream>>>(hs, Wv, Vf, M, NKV * Dd, Hh);

    const int totQ4 = Bb * Ss * NH  * Dd / 4;        // 2097152
    const int totK4 = Bb * Ss * NKV * Dd / 4;        //  524288
    rope_pack4<<<totQ4 / 256, 256, 0, stream>>>(Qf, cosT, sinT, Qh, NH, 1, totQ4);
    rope_pack4<<<totK4 / 256, 256, 0, stream>>>(Kf, cosT, sinT, Kh, NKV, 1, totK4);
    rope_pack4<<<totK4 / 256, 256, 0, stream>>>(Vf, cosT, sinT, Vh, NKV, 0, totK4);

    flash_attn_wmma<<<Bb * NH * (Ss / 64), blk, 0, stream>>>(Qh, Kh, Vh, Of);

    gemm_xw_wmma<<<dim3(Hh / 64, M / 64), blk, 0, stream>>>(Of, Wo, out, M, Hh, Hh);
}
